// TransformerBlock_5987184411264
// MI455X (gfx1250) — matrix operand order repaired and checked by reading, not compiled, not run on hardware
//
#include <hip/hip_runtime.h>

// ---------------------------------------------------------------------------
// CDNA5 (gfx1250) TransformerBlock: bf16 WMMA GEMMs + flash attention.
// D=256, H=4, B=32, L=512. wave32, v_wmma_f32_16x16x32_bf16 throughout.
// K=64 staged slabs (8 WMMA per barrier) + async global->LDS staging
// (ASYNCcnt) via __builtin_amdgcn_global_load_async_to_lds_b64.
// ---------------------------------------------------------------------------

typedef __attribute__((ext_vector_type(16))) __bf16 v16bf;
typedef __attribute__((ext_vector_type(8)))  float  v8f;
typedef int v2i __attribute__((ext_vector_type(2)));

#define NEGV (-4294967296.0f)   // float(-(2^32)+1)

#if defined(__gfx1250__) &&                                             \
    __has_builtin(__builtin_amdgcn_global_load_async_to_lds_b64) &&     \
    __has_builtin(__builtin_amdgcn_s_wait_asynccnt)
#define USE_ASYNC_LDS 1
#else
#define USE_ASYNC_LDS 0
#endif

#if USE_ASYNC_LDS
static __device__ __forceinline__ void async_copy_b64(const void* g, void* l) {
  // GLOBAL_LOAD_ASYNC_TO_LDS_B64: per-lane 8B memory -> LDS, tracked ASYNCcnt.
  // Builtin signature uses v2i* (2 x i32) pointers: global src, LDS dst.
  __builtin_amdgcn_global_load_async_to_lds_b64(
      (__attribute__((address_space(1))) v2i*)g,
      (__attribute__((address_space(3))) v2i*)l, 0, 0);
}
static __device__ __forceinline__ void async_wait0() {
  __builtin_amdgcn_s_wait_asynccnt(0);
}
#endif

static __device__ __forceinline__ unsigned short f2bf(float f) {
  unsigned int u = __float_as_uint(f);
  u += 0x7FFFu + ((u >> 16) & 1u);          // round-to-nearest-even
  return (unsigned short)(u >> 16);
}

static __device__ __forceinline__ v8f wmma_bf16(v16bf a, v16bf b, v8f c) {
  // (neg_a, A, neg_b, B, c_mod, C, reuse_a, reuse_b)
  return __builtin_amdgcn_wmma_f32_16x16x32_bf16(false, a, false, b, (short)0, c,
                                                 false, false);
}

// A-fragment (16x32 bf16, M x K).  Lane layout per CDNA5 ISA 7.12.2:
//   m = lane&15, half = lane>>4; VGPR i holds K pair at
//   kk = (i<4?0:16) + 8*half + 2*(i&3).
// Works on either global or LDS pointers (addrspace inferred after inline).
static __device__ __forceinline__ v16bf load_a_frag(const unsigned short* base,
                                                    int stride) {
  const int lane = threadIdx.x & 31;
  const int half = lane >> 4, m = lane & 15;
  union { v16bf v; unsigned int u[8]; } r;
#pragma unroll
  for (int i = 0; i < 8; ++i) {
    const int kk = ((i & 4) << 2) + (half << 3) + ((i & 3) << 1);
    r.u[i] = *(const unsigned int*)(base + m * stride + kk);
  }
  return r.v;
}

// B-fragment (32x16 bf16, K x N): n = lane&15, element s holds
//   k = (s<8?0:16) + 8*half + (s&7)  (same K order as A).
static __device__ __forceinline__ v16bf load_b_frag(const unsigned short* base,
                                                    int stride) {
  const int lane = threadIdx.x & 31;
  const int half = lane >> 4, n = lane & 15;
  union { v16bf v; unsigned short s[16]; } r;
#pragma unroll
  for (int i = 0; i < 16; ++i) {
    const int kk = ((i & 8) << 1) + (half << 3) + (i & 7);
    r.s[i] = base[kk * stride + n];
  }
  return r.v;
}

// ---------------------------------------------------------------------------
// Small prep kernels
// ---------------------------------------------------------------------------
__global__ void cvt_bf16(const float* __restrict__ in,
                         unsigned short* __restrict__ out, int n) {
  for (int i = blockIdx.x * blockDim.x + threadIdx.x; i < n;
       i += gridDim.x * blockDim.x)
    out[i] = f2bf(in[i]);
}

// out[k*256 + j] = bf16(in[j*256 + k])   (256x256 transpose for w1/w2: X @ W^T)
__global__ void cvt_bf16_T(const float* __restrict__ in,
                           unsigned short* __restrict__ out) {
  for (int i = blockIdx.x * blockDim.x + threadIdx.x; i < 65536;
       i += gridDim.x * blockDim.x) {
    const int j = i >> 8, k = i & 255;
    out[k * 256 + j] = f2bf(in[i]);
  }
}

// Qp = Q + pe  (kept in f32 as the attention residual)
__global__ void add_pe_f32(const float* __restrict__ X,
                           const float* __restrict__ pe,
                           float* __restrict__ out, int n) {
  for (int i = blockIdx.x * blockDim.x + threadIdx.x; i < n;
       i += gridDim.x * blockDim.x) {
    const int d = i & 255;
    const int l = (i >> 8) & 511;
    out[i] = X[i] + pe[l * 256 + d];
  }
}

// ---------------------------------------------------------------------------
// Tiled bf16 GEMM.  C(MxN) = A(MxK) @ B(KxN).  K must be a multiple of 64.
//   a_f32_pe : A is f32 with positional encoding fused in (lda=K=256)
//   mode 0   : bf16 row-major store (+bias/relu optional)
//   mode 1   : KhT store: C[row=b*512+l][col=h*256+d] -> KhT[(b*4+h)][d][l]
//   mode 2   : f32 store with residual add (+bias optional)
// Block: 256 threads (8 waves); tile 128(M) x 64(N); wave tile 32x32.
// K staged 64-deep -> 8 WMMA per workgroup barrier.
// ---------------------------------------------------------------------------
__global__ void __launch_bounds__(256)
gemm_bf16_kernel(const void* __restrict__ Aptr,
                 const unsigned short* __restrict__ Bw,
                 const float* __restrict__ pe, void* __restrict__ Cout,
                 const float* __restrict__ bias,
                 const float* __restrict__ resid, int M, int N, int K,
                 int mode, int relu, int a_f32_pe) {
  __shared__ unsigned short As[128 * 72];  // 64 cols padded to 72 (bank-safe)
  __shared__ unsigned short Bs[64 * 72];

  const int tid = threadIdx.x;
  const int lane = tid & 31, wave = tid >> 5;
  const int mw = wave >> 1, nw = wave & 1;
  const int row0 = blockIdx.y * 128, col0 = blockIdx.x * 64;

  union { v8f v; float f[8]; } acc[2][2];
#pragma unroll
  for (int a = 0; a < 2; ++a)
#pragma unroll
    for (int bI = 0; bI < 2; ++bI)
#pragma unroll
      for (int r = 0; r < 8; ++r) acc[a][bI].f[r] = 0.0f;

  for (int k0 = 0; k0 < K; k0 += 64) {
    __syncthreads();
    // ---- stage A tile 128x64 ----
    if (a_f32_pe) {
      const float* Af = (const float*)Aptr;
#pragma unroll
      for (int e = tid; e < 4096; e += 256) {   // f32 pairs -> packed bf16
        const int r = e >> 5, c2 = (e & 31) << 1;
        const int grow = row0 + r, gc = k0 + c2;
        const float* src = Af + (size_t)grow * K + gc;
        const float* pp = pe + (size_t)(grow & 511) * 256 + gc;
        const unsigned int pk =
            (unsigned int)f2bf(src[0] + pp[0]) |
            ((unsigned int)f2bf(src[1] + pp[1]) << 16);
        *(unsigned int*)&As[r * 72 + c2] = pk;
      }
    } else {
      const unsigned short* Ab = (const unsigned short*)Aptr;
#pragma unroll
      for (int e = tid; e < 2048; e += 256) {   // 8B per lane
        const int r = e >> 4, seg = (e & 15) << 2;
#if USE_ASYNC_LDS
        async_copy_b64(Ab + (size_t)(row0 + r) * K + k0 + seg,
                       &As[r * 72 + seg]);
#else
        *(uint2*)&As[r * 72 + seg] =
            *(const uint2*)(Ab + (size_t)(row0 + r) * K + k0 + seg);
#endif
      }
    }
    // ---- stage B tile 64x64 + prefetch next K slab ----
#pragma unroll
    for (int e = tid; e < 1024; e += 256) {
      const int r = e >> 4, seg = (e & 15) << 2;
#if USE_ASYNC_LDS
      async_copy_b64(Bw + (size_t)(k0 + r) * N + col0 + seg,
                     &Bs[r * 72 + seg]);
#else
      *(uint2*)&Bs[r * 72 + seg] =
          *(const uint2*)(Bw + (size_t)(k0 + r) * N + col0 + seg);
#endif
      if (k0 + 64 < K)
        __builtin_prefetch(Bw + (size_t)(k0 + 64 + r) * N + col0 + seg, 0, 1);
    }
#if USE_ASYNC_LDS
    async_wait0();
#endif
    __syncthreads();

#pragma unroll
    for (int ks = 0; ks < 2; ++ks) {
      const v16bf a0 = load_a_frag(As + (mw * 32) * 72 + ks * 32, 72);
      const v16bf a1 = load_a_frag(As + (mw * 32 + 16) * 72 + ks * 32, 72);
      const v16bf b0 = load_b_frag(Bs + (ks * 32) * 72 + nw * 32, 72);
      const v16bf b1 = load_b_frag(Bs + (ks * 32) * 72 + nw * 32 + 16, 72);
      acc[0][0].v = wmma_bf16(a0, b0, acc[0][0].v);
      acc[0][1].v = wmma_bf16(a0, b1, acc[0][1].v);
      acc[1][0].v = wmma_bf16(a1, b0, acc[1][0].v);
      acc[1][1].v = wmma_bf16(a1, b1, acc[1][1].v);
    }
  }

  const int n = lane & 15, half = lane >> 4;
  if (mode == 1) {
    // per-head transposed store: contiguous 8 bf16 along l -> one uint4
#pragma unroll
    for (int sm = 0; sm < 2; ++sm)
#pragma unroll
      for (int sn = 0; sn < 2; ++sn) {
        const int gcol = col0 + nw * 32 + sn * 16 + n;
        const int h = gcol >> 8, d = gcol & 255;
        const int growb = row0 + mw * 32 + sm * 16 + half * 8;
        const int b = growb >> 9, l0 = growb & 511;
        unsigned short* dst = (unsigned short*)Cout +
                              ((size_t)(b * 4 + h) << 17) + ((size_t)d << 9) +
                              l0;
        union { uint4 q; unsigned short s[8]; } pk;
#pragma unroll
        for (int r = 0; r < 8; ++r) pk.s[r] = f2bf(acc[sm][sn].f[r]);
        *(uint4*)dst = pk.q;
      }
  } else {
#pragma unroll
    for (int sm = 0; sm < 2; ++sm)
#pragma unroll
      for (int sn = 0; sn < 2; ++sn) {
        const int gcol = col0 + nw * 32 + sn * 16 + n;
        const float bv = bias ? bias[gcol] : 0.0f;
#pragma unroll
        for (int r = 0; r < 8; ++r) {
          const int grow = row0 + mw * 32 + sm * 16 + half * 8 + r;
          float v = acc[sm][sn].f[r] + bv;
          if (relu) v = fmaxf(v, 0.0f);
          const size_t idx = (size_t)grow * N + gcol;
          if (mode == 2)
            ((float*)Cout)[idx] = v + (resid ? resid[idx] : 0.0f);
          else
            ((unsigned short*)Cout)[idx] = f2bf(v);
        }
      }
  }
}

// ---------------------------------------------------------------------------
// Flash attention: grid (128 bh, 8 qblocks), 128 threads (4 waves).
// Each wave: 16 queries, full 256-d f32 accumulator (128 VGPRs), keys in
// blocks of 32 with causal early exit + online softmax.
// ---------------------------------------------------------------------------
__global__ void __launch_bounds__(128, 1)
attn_kernel(const unsigned short* __restrict__ Qh,
            const unsigned short* __restrict__ KhT,
            const unsigned short* __restrict__ Vh,
            const unsigned char* __restrict__ maskp,
            unsigned short* __restrict__ Vatt) {
  __shared__ unsigned short Ks[256 * 40];   // [d][32 keys], pad 40
  __shared__ unsigned short Vs[32 * 268];   // [key][256 d], pad 268
  __shared__ unsigned short Ps[4][16 * 34]; // per-wave P tile, pad 34

  const int tid = threadIdx.x, lane = tid & 31, wave = tid >> 5;
  const int bh = blockIdx.x, b = bh >> 2, h = bh & 3;
  const int qb = blockIdx.y;
  const int q0 = qb * 64 + wave * 16;
  const int n = lane & 15, half = lane >> 4;

  // preload this wave's Q A-fragments (16 x 256 = 8 frags) into registers
  v16bf qf[8];
  {
    const unsigned short* qbase =
        Qh + ((size_t)b * 512 + q0) * 1024 + h * 256;
#pragma unroll
    for (int ks = 0; ks < 8; ++ks) qf[ks] = load_a_frag(qbase + ks * 32, 1024);
  }

  union { v8f v; float f[8]; } acc[16];
#pragma unroll
  for (int t = 0; t < 16; ++t)
#pragma unroll
    for (int r = 0; r < 8; ++r) acc[t].f[r] = 0.0f;

  float mrun[8], lrun[8];
  unsigned char rmask[8];
#pragma unroll
  for (int r = 0; r < 8; ++r) {
    mrun[r] = -3.0e38f;
    lrun[r] = 0.0f;
    rmask[r] = maskp[b * 512 + q0 + half * 8 + r];
  }

  const int kend = qb * 64 + 63;                 // WG-uniform causal bound
  const size_t kbase = (size_t)bh << 17;         // KhT per-bh (256*512)
  const size_t vbase = (size_t)b * 524288 + (size_t)h * 256;
  const float inv_scale = 1.0f / (16.0f + 1e-6f);

  for (int k0 = 0; k0 <= kend; k0 += 32) {
    __syncthreads();
    // stage K^T block: 256 rows (d) x 32 keys  (8B per lane per op)
#pragma unroll
    for (int e = tid; e < 2048; e += 128) {
      const int d = e >> 3, seg = (e & 7) << 2;
#if USE_ASYNC_LDS
      async_copy_b64(KhT + kbase + (size_t)d * 512 + k0 + seg,
                     &Ks[d * 40 + seg]);
#else
      *(uint2*)&Ks[d * 40 + seg] =
          *(const uint2*)(KhT + kbase + (size_t)d * 512 + k0 + seg);
#endif
    }
    // stage V block: 32 keys x 256 d
#pragma unroll
    for (int e = tid; e < 2048; e += 128) {
      const int kk = e >> 6, seg = (e & 63) << 2;
#if USE_ASYNC_LDS
      async_copy_b64(Vh + vbase + (size_t)(k0 + kk) * 1024 + seg,
                     &Vs[kk * 268 + seg]);
#else
      *(uint2*)&Vs[kk * 268 + seg] =
          *(const uint2*)(Vh + vbase + (size_t)(k0 + kk) * 1024 + seg);
#endif
    }
#if USE_ASYNC_LDS
    async_wait0();
#endif
    __syncthreads();
    if (k0 > q0 + 15) continue;  // wave-uniform: this wave is causally done

    // ---- S = Q . K^T  (16 x 32) ----
    union { v8f v; float f[8]; } s0, s1;
#pragma unroll
    for (int r = 0; r < 8; ++r) { s0.f[r] = 0.0f; s1.f[r] = 0.0f; }
#pragma unroll
    for (int ks = 0; ks < 8; ++ks) {
      const v16bf kb0 = load_b_frag(Ks + (ks * 32) * 40, 40);
      const v16bf kb1 = load_b_frag(Ks + (ks * 32) * 40 + 16, 40);
      s0.v = wmma_bf16(qf[ks], kb0, s0.v);
      s1.v = wmma_bf16(qf[ks], kb1, s1.v);
    }

    // ---- scale + mask + online softmax, build P tile in LDS ----
#pragma unroll
    for (int r = 0; r < 8; ++r) {
      const int qrow = q0 + half * 8 + r;
      float v0 = s0.f[r] * inv_scale;
      float v1 = s1.f[r] * inv_scale;
      if (rmask[r] || (k0 + n) > qrow) v0 = NEGV;
      if (rmask[r] || (k0 + 16 + n) > qrow) v1 = NEGV;
      float rmax = fmaxf(v0, v1);
      rmax = fmaxf(rmax, __shfl_xor(rmax, 1, 16));
      rmax = fmaxf(rmax, __shfl_xor(rmax, 2, 16));
      rmax = fmaxf(rmax, __shfl_xor(rmax, 4, 16));
      rmax = fmaxf(rmax, __shfl_xor(rmax, 8, 16));
      const float mnew = fmaxf(mrun[r], rmax);
      const float esc = __expf(mrun[r] - mnew);
      const float p0 = __expf(v0 - mnew);
      const float p1 = __expf(v1 - mnew);
      float ps = p0 + p1;
      ps += __shfl_xor(ps, 1, 16);
      ps += __shfl_xor(ps, 2, 16);
      ps += __shfl_xor(ps, 4, 16);
      ps += __shfl_xor(ps, 8, 16);
      lrun[r] = lrun[r] * esc + ps;
      mrun[r] = mnew;
#pragma unroll
      for (int t = 0; t < 16; ++t) acc[t].f[r] *= esc;
      Ps[wave][(half * 8 + r) * 34 + n] = f2bf(p0);
      Ps[wave][(half * 8 + r) * 34 + 16 + n] = f2bf(p1);
    }

    // ---- acc += P . V  (same-wave LDS round trip re-swizzles C->A layout)
    const v16bf pa = load_a_frag(Ps[wave], 34);
#pragma unroll
    for (int t = 0; t < 16; ++t) {
      const v16bf bv = load_b_frag(Vs + t * 16, 268);
      acc[t].v = wmma_bf16(pa, bv, acc[t].v);
    }
  }

  // ---- normalize + store Vatt (b, l, h, d) bf16 ----
  float rinv[8];
#pragma unroll
  for (int r = 0; r < 8; ++r) rinv[r] = 1.0f / lrun[r];
  unsigned short* outp = Vatt + (size_t)b * 524288 + (size_t)h * 256 + n;
#pragma unroll
  for (int t = 0; t < 16; ++t)
#pragma unroll
    for (int r = 0; r < 8; ++r) {
      const int qrow = q0 + half * 8 + r;
      outp[(size_t)qrow * 1024 + t * 16] = f2bf(acc[t].f[r] * rinv[r]);
    }
}

// ---------------------------------------------------------------------------
// Row LayerNorm (row = 256 = one block of 256 threads, 8 waves).
// ---------------------------------------------------------------------------
__global__ void __launch_bounds__(256)
ln_kernel(const float* __restrict__ X, const float* __restrict__ gamma,
          const float* __restrict__ beta, float* __restrict__ outf,
          unsigned short* __restrict__ outb) {
  const int row = blockIdx.x, t = threadIdx.x;
  const int lane = t & 31, w = t >> 5;
  __shared__ float red[8];
  const float v = X[(size_t)row * 256 + t];

  float s = v;
#pragma unroll
  for (int m = 1; m < 32; m <<= 1) s += __shfl_xor(s, m, 32);
  if (lane == 0) red[w] = s;
  __syncthreads();
  float mu = 0.0f;
#pragma unroll
  for (int i = 0; i < 8; ++i) mu += red[i];
  mu *= (1.0f / 256.0f);
  __syncthreads();

  const float d = v - mu;
  float s2 = d * d;
#pragma unroll
  for (int m = 1; m < 32; m <<= 1) s2 += __shfl_xor(s2, m, 32);
  if (lane == 0) red[w] = s2;
  __syncthreads();
  float var = 0.0f;
#pragma unroll
  for (int i = 0; i < 8; ++i) var += red[i];
  var *= (1.0f / 256.0f);

  const float y = d * rsqrtf(var + 1e-5f) * gamma[t] + beta[t];
  outf[(size_t)row * 256 + t] = y;
  if (outb) outb[(size_t)row * 256 + t] = f2bf(y);
}

// ---------------------------------------------------------------------------
extern "C" void kernel_launch(void* const* d_in, const int* in_sizes, int n_in,
                              void* d_out, int out_size, void* d_ws,
                              size_t ws_size, hipStream_t stream) {
  (void)in_sizes; (void)n_in; (void)out_size; (void)ws_size;
  const float* Q = (const float*)d_in[0];
  const float* Kin = (const float*)d_in[1];
  const float* Vin = (const float*)d_in[2];
  const unsigned char* mask = (const unsigned char*)d_in[3];
  const float* pe = (const float*)d_in[4];
  const float* W_q = (const float*)d_in[5];
  const float* W_k = (const float*)d_in[6];
  const float* W_v = (const float*)d_in[7];
  const float* W_o = (const float*)d_in[8];
  const float* w1 = (const float*)d_in[9];
  const float* b1 = (const float*)d_in[10];
  const float* w2 = (const float*)d_in[11];
  const float* b2 = (const float*)d_in[12];
  const float* gamma = (const float*)d_in[13];
  const float* beta = (const float*)d_in[14];

  char* ws = (char*)d_ws;
  size_t off = 0;
  auto alloc = [&](size_t bytes) -> char* {
    char* p = ws + off;
    off += (bytes + 255) & ~(size_t)255;
    return p;
  };
  const size_t ML = (size_t)16384;  // B*L rows
  unsigned short* Wq_b = (unsigned short*)alloc(262144u * 2);
  unsigned short* Wk_b = (unsigned short*)alloc(262144u * 2);
  unsigned short* Wv_b = (unsigned short*)alloc(262144u * 2);
  unsigned short* Wo_b = (unsigned short*)alloc(262144u * 2);
  unsigned short* W1t = (unsigned short*)alloc(65536u * 2);
  unsigned short* W2t = (unsigned short*)alloc(65536u * 2);
  float* Qp = (float*)alloc(ML * 256 * 4);
  unsigned short* Qh = (unsigned short*)alloc(ML * 1024 * 2);
  unsigned short* KhT = (unsigned short*)alloc(ML * 1024 * 2);
  unsigned short* Vh = (unsigned short*)alloc(ML * 1024 * 2);
  unsigned short* Vatt = (unsigned short*)alloc(ML * 1024 * 2);
  float* X0 = (float*)alloc(ML * 256 * 4);
  float* Xf = (float*)alloc(ML * 256 * 4);
  unsigned short* Xb = (unsigned short*)alloc(ML * 256 * 2);
  unsigned short* H1 = (unsigned short*)alloc(ML * 256 * 2);
  float* X1 = (float*)alloc(ML * 256 * 4);

  // weight prep
  cvt_bf16<<<256, 256, 0, stream>>>(W_q, Wq_b, 262144);
  cvt_bf16<<<256, 256, 0, stream>>>(W_k, Wk_b, 262144);
  cvt_bf16<<<256, 256, 0, stream>>>(W_v, Wv_b, 262144);
  cvt_bf16<<<256, 256, 0, stream>>>(W_o, Wo_b, 262144);
  cvt_bf16_T<<<256, 256, 0, stream>>>(w1, W1t);
  cvt_bf16_T<<<256, 256, 0, stream>>>(w2, W2t);
  add_pe_f32<<<512, 256, 0, stream>>>(Q, pe, Qp, 16384 * 256);

  // projections (PE-add fused into A staging)
  dim3 gp(1024 / 64, 16384 / 128);
  gemm_bf16_kernel<<<gp, 256, 0, stream>>>(Q, Wq_b, pe, Qh, nullptr, nullptr,
                                           16384, 1024, 256, 0, 0, 1);
  gemm_bf16_kernel<<<gp, 256, 0, stream>>>(Kin, Wk_b, pe, KhT, nullptr,
                                           nullptr, 16384, 1024, 256, 1, 0, 1);
  gemm_bf16_kernel<<<gp, 256, 0, stream>>>(Vin, Wv_b, pe, Vh, nullptr, nullptr,
                                           16384, 1024, 256, 0, 0, 1);

  // flash attention
  attn_kernel<<<dim3(128, 8), 128, 0, stream>>>(Qh, KhT, Vh, mask, Vatt);

  // out-proj (+Qp residual), LN1, FFN1 (relu), FFN2 (+X residual), LN2
  dim3 go(256 / 64, 16384 / 128);
  gemm_bf16_kernel<<<go, 256, 0, stream>>>(Vatt, Wo_b, nullptr, X0, nullptr,
                                           Qp, 16384, 256, 1024, 2, 0, 0);
  ln_kernel<<<16384, 256, 0, stream>>>(X0, gamma, beta, Xf, Xb);
  gemm_bf16_kernel<<<go, 256, 0, stream>>>(Xb, W1t, nullptr, H1, b1, nullptr,
                                           16384, 256, 256, 0, 1, 0);
  gemm_bf16_kernel<<<go, 256, 0, stream>>>(H1, W2t, nullptr, X1, b2, Xf,
                                           16384, 256, 256, 2, 0, 0);
  ln_kernel<<<16384, 256, 0, stream>>>(X1, gamma, beta, (float*)d_out, nullptr);
}
